// SparseAutoencoder_87273735454790
// MI455X (gfx1250) — compile-verified, matrix operand order checked by reading, not run on hardware
//
#include <hip/hip_runtime.h>

// ---------------------------------------------------------------------------
// Problem constants (reference: B=1024, H=4096, D=65536, k=64)
// ---------------------------------------------------------------------------
#define BB 1024
#define HH 4096
#define DD 65536
#define KK 64

typedef __attribute__((ext_vector_type(16))) __bf16 v16bf;
typedef __attribute__((ext_vector_type(8)))  __bf16 v8bf;
typedef __attribute__((ext_vector_type(8)))  float  v8f;
typedef __attribute__((ext_vector_type(4)))  unsigned int u32x4;
typedef __attribute__((ext_vector_type(8)))  int    i32x8;
typedef __attribute__((ext_vector_type(4)))  int    i32x4;

#define LDS_STRIDE 40  // shorts per LDS tile row (32 data + 8 pad); 80B, 16B-aligned

// Convert 16 f32 -> 16 bf16, store as two 16B LDS writes
__device__ inline void store16_bf16(unsigned short* dst, const float* s) {
    v8bf a, b;
#pragma unroll
    for (int i = 0; i < 8; ++i) {
        a[i] = (__bf16)s[i];
        b[i] = (__bf16)s[8 + i];
    }
    *(u32x4*)dst       = __builtin_bit_cast(u32x4, a);
    *(u32x4*)(dst + 8) = __builtin_bit_cast(u32x4, b);
}

// 16 consecutive f32 from global into registers (4x global_load_b128)
__device__ inline void gload16(float* r, const float* p) {
    const float4* p4 = (const float4*)p;
#pragma unroll
    for (int q = 0; q < 4; ++q) {
        float4 v = p4[q];
        r[4 * q + 0] = v.x; r[4 * q + 1] = v.y;
        r[4 * q + 2] = v.z; r[4 * q + 3] = v.w;
    }
}

// Load a 16x32 bf16 fragment (A-layout; B uses same layout on [n][k] storage).
// lane<16 gets K=0..7,16..23 of row; lane>=16 gets K=8..15,24..31.
__device__ inline v16bf load_frag(const unsigned short* base, int row, int half) {
    const unsigned short* p = base + row * LDS_STRIDE + half * 8;
    union { u32x4 q[2]; v16bf v; } u;
    u.q[0] = *(const u32x4*)p;
    u.q[1] = *(const u32x4*)(p + 16);
    return u.v;
}

// ---------------------------------------------------------------------------
// Kernel 1: encoder GEMM  pre_act[b,d] = sum_h (x[b,h]-pre_bias[h])*W_enc[d,h] + b_enc[d]
// Block tile 128(M) x 128(N), K-step 32, 8 waves (2M x 4N), wave tile 64x32.
// Double-buffered LDS + register prefetch: global loads for tile k+1 issue
// before the WMMAs of tile k, so HBM/L2 latency hides under the matrix ops.
// grid ordered M-fastest so the 8 blocks sharing a W_enc slice are adjacent.
// ---------------------------------------------------------------------------
__global__ __launch_bounds__(256)
void sae_encoder_wmma(const float* __restrict__ X,
                      const float* __restrict__ Wenc,
                      const float* __restrict__ benc,
                      const float* __restrict__ prebias,
                      float* __restrict__ preact) {
    __shared__ __attribute__((aligned(16))) unsigned short Abuf[2][128 * LDS_STRIDE];
    __shared__ __attribute__((aligned(16))) unsigned short Bbuf[2][128 * LDS_STRIDE];

    const int t   = threadIdx.x;
    const int bid = blockIdx.x;
    const int m0  = (bid & 7) * 128;   // M-tile fastest -> W_enc slice shared by adjacent blocks
    const int n0  = (bid >> 3) * 128;

    const int lane = t & 31;
    const int wid  = t >> 5;
    const int wm   = (wid & 1) * 64;   // wave M offset within tile
    const int wn   = (wid >> 1) * 32;  // wave N offset within tile
    const int r16  = lane & 15;
    const int half = lane >> 4;

    const int srow = t >> 1;           // staging: row handled by this thread
    const int seg  = t & 1;            // which 16-element K segment
    const int soff = srow * LDS_STRIDE + seg * 16;

    const float* Xp = X + (size_t)(m0 + srow) * HH + seg * 16;
    const float* Wp = Wenc + (size_t)(n0 + srow) * HH + seg * 16;
    const float* Pp = prebias + seg * 16;

    float ra[16], rb[16], rp[16], tmp[16];

    // Prologue: stage tile 0
    gload16(ra, Xp);
    gload16(rb, Wp);
    gload16(rp, Pp);
#pragma unroll
    for (int i = 0; i < 16; ++i) tmp[i] = ra[i] - rp[i];
    store16_bf16(&Abuf[0][soff], tmp);
    store16_bf16(&Bbuf[0][soff], rb);

    v8f c[4][2] = {};
    int cur = 0;

    for (int it = 0; it < HH / 32; ++it) {
        __syncthreads();               // buf[cur] published; buf[cur^1] drained

        const bool more = (it + 1 < HH / 32);
        if (more) {                    // issue next tile's global loads NOW
            const int kn = (it + 1) * 32;
            gload16(ra, Xp + kn);
            gload16(rb, Wp + kn);
            gload16(rp, Pp + kn);
        }

        // ---- WMMA: 4 M-frags x 2 N-frags per wave on buf[cur] ----
        v16bf bf[2];
#pragma unroll
        for (int ni = 0; ni < 2; ++ni)
            bf[ni] = load_frag(&Bbuf[cur][0], wn + ni * 16 + r16, half);
#pragma unroll
        for (int mi = 0; mi < 4; ++mi) {
            v16bf af = load_frag(&Abuf[cur][0], wm + mi * 16 + r16, half);
#pragma unroll
            for (int ni = 0; ni < 2; ++ni)
                c[mi][ni] = __builtin_amdgcn_wmma_f32_16x16x32_bf16(
                    false, af, false, bf[ni], (short)0, c[mi][ni], false, false);
        }

        if (more) {                    // convert + stage into the other buffer
#pragma unroll
            for (int i = 0; i < 16; ++i) tmp[i] = ra[i] - rp[i];
            store16_bf16(&Abuf[cur ^ 1][soff], tmp);
            store16_bf16(&Bbuf[cur ^ 1][soff], rb);
        }
        cur ^= 1;
    }

    // ---- epilogue: add b_enc, store C (VGPR r: M=r or r+8 by lane half, N=lane&15) ----
#pragma unroll
    for (int ni = 0; ni < 2; ++ni) {
        const int gn = n0 + wn + ni * 16 + r16;
        const float bias = benc[gn];
#pragma unroll
        for (int mi = 0; mi < 4; ++mi) {
#pragma unroll
            for (int r = 0; r < 8; ++r) {
                const int gm = m0 + wm + mi * 16 + r + half * 8;
                preact[(size_t)gm * DD + gn] = c[mi][ni][r] + bias;
            }
        }
    }
}

// ---------------------------------------------------------------------------
// Kernel 2: per-row top-64 (tournament). One block of 1024 threads per row.
// Each thread owns the 64 elements {t + 1024*j}; 64 rounds of argmax reduce,
// winner rescans its chunk using a 64-bit removal mask.
// ---------------------------------------------------------------------------
__global__ __launch_bounds__(1024)
void sae_topk(const float* __restrict__ preact,
              float* __restrict__ topv,
              float* __restrict__ topi) {
    __shared__ float sv[1024];
    __shared__ int   si[1024];
    __shared__ float sv2[256];
    __shared__ int   si2[256];
    __shared__ float gvS;
    __shared__ int   giS;

    const int t = threadIdx.x;
    const int b = blockIdx.x;
    const float* row = preact + (size_t)b * DD;

    const float NEG = -3.402823e38f;
    float lv = NEG; int li = 0;
    unsigned long long removed = 0ULL;
#pragma unroll 4
    for (int j = 0; j < 64; ++j) {
        float v = row[j * 1024 + t];
        if (v > lv) { lv = v; li = j * 1024 + t; }
    }
    sv[t] = lv; si[t] = li;

    for (int i = 0; i < KK; ++i) {
        __syncthreads();
        if (t < 256) {
            float m = sv[t * 4]; int id = si[t * 4];
#pragma unroll
            for (int q = 1; q < 4; ++q) {
                float v = sv[t * 4 + q]; int ii = si[t * 4 + q];
                if (v > m || (v == m && ii < id)) { m = v; id = ii; }
            }
            sv2[t] = m; si2[t] = id;
        }
        __syncthreads();
        if (t < 32) {
            float m = sv2[t * 8]; int id = si2[t * 8];
#pragma unroll
            for (int q = 1; q < 8; ++q) {
                float v = sv2[t * 8 + q]; int ii = si2[t * 8 + q];
                if (v > m || (v == m && ii < id)) { m = v; id = ii; }
            }
#pragma unroll
            for (int off = 16; off > 0; off >>= 1) {
                float ov = __shfl_xor(m, off, 32);
                int   oi = __shfl_xor(id, off, 32);
                if (ov > m || (ov == m && oi < id)) { m = ov; id = oi; }
            }
            if (t == 0) { gvS = m; giS = id; }
        }
        __syncthreads();
        const float gv = gvS;
        const int   gi = giS;
        if (t == 0) {
            topv[b * KK + i] = gv > 0.f ? gv : 0.f;   // relu on kept values
            topi[b * KK + i] = (float)gi;
        }
        if ((gi & 1023) == t) {                       // owner rescans its chunk
            removed |= 1ULL << (gi >> 10);
            float m = NEG; int id = 0;
            for (int j = 0; j < 64; ++j) {
                if ((removed >> j) & 1ULL) continue;
                float v = row[j * 1024 + t];
                if (v > m) { m = v; id = j * 1024 + t; }
            }
            sv[t] = m; si[t] = id;
        }
    }
}

// ---------------------------------------------------------------------------
// Kernel 3: decoder, transposed. One block per output feature h (4096 blocks).
// The 256KB W_dec row is DMA'd into LDS by the Tensor Data Mover (one wave
// issues tensor_load_to_lds, waits on TENSORcnt, then the workgroup barrier
// publishes it). D#: 2D tile 16384x4 elements, data_size=4B, stride=16384.
// W_dec is read from HBM exactly once, coalesced, with zero VGPR staging.
// ---------------------------------------------------------------------------
__global__ __launch_bounds__(256)
void sae_decoder(const float* __restrict__ Wdec,
                 const float* __restrict__ prebias,
                 const float* __restrict__ topv,
                 const float* __restrict__ topi,
                 float* __restrict__ xhat) {
    extern __shared__ float wrow[];    // DD floats = 256 KB
    const int h = blockIdx.x;
    const int t = threadIdx.x;
    const float* src = Wdec + (size_t)h * DD;

    if (t < 32) {                      // one wave issues the TDM descriptor
        unsigned long long ga = (unsigned long long)(const void*)src;
        unsigned int ldsoff = (unsigned int)(unsigned long long)(void*)wrow;
        // D# group0: count=1 | lds_addr | global_addr[56:0] | type=2
        u32x4 g0 = { 1u,
                     ldsoff,
                     (unsigned int)ga,
                     (unsigned int)((ga >> 32) & 0x01FFFFFFull) | (2u << 30) };
        // D# group1: data_size=4B; tensor_dim0=16384; tensor_dim1=4;
        //            tile_dim0=16384; tile_dim1=4; tensor_dim0_stride=16384
        i32x8 g1 = { (int)(2u << 16),        // [17:16] data_size=2 (4 bytes)
                     (int)(16384u << 16),    // tensor_dim0[15:0] in [31:16]
                     (int)(4u << 16),        // tensor_dim0 hi=0 | tensor_dim1 lo=4
                     (int)(16384u << 16),    // tensor_dim1 hi=0 | tile_dim0=16384
                     4,                      // tile_dim1=4, tile_dim2=0
                     16384,                  // tensor_dim0_stride lo
                     0, 0 };
        i32x4 z4 = { 0, 0, 0, 0 };
#if __clang_major__ >= 23
        i32x8 z8 = { 0, 0, 0, 0, 0, 0, 0, 0 };
        __builtin_amdgcn_tensor_load_to_lds(g0, g1, z4, z4, z8, 0);
#else
        __builtin_amdgcn_tensor_load_to_lds(g0, g1, z4, z4, 0);
#endif
        __builtin_amdgcn_s_wait_tensorcnt(0);
    }
    __syncthreads();

    const float pb = prebias[h];
#pragma unroll
    for (int bb = 0; bb < 4; ++bb) {
        const int b = bb * 256 + t;
        const float4* v4 = (const float4*)(topv + b * KK);
        const float4* i4 = (const float4*)(topi + b * KK);
        float acc = pb;
#pragma unroll
        for (int c = 0; c < KK / 4; ++c) {
            float4 v = v4[c]; float4 ix = i4[c];
            acc += v.x * wrow[(int)ix.x];
            acc += v.y * wrow[(int)ix.y];
            acc += v.z * wrow[(int)ix.z];
            acc += v.w * wrow[(int)ix.w];
        }
        xhat[(size_t)b * HH + h] = acc;
    }
}

// ---------------------------------------------------------------------------
// Host launcher. d_out layout: [x_hat | top_vals | top_idx | pre_act] (flat f32).
// ---------------------------------------------------------------------------
extern "C" void kernel_launch(void* const* d_in, const int* in_sizes, int n_in,
                              void* d_out, int out_size, void* d_ws, size_t ws_size,
                              hipStream_t stream) {
    (void)in_sizes; (void)n_in; (void)out_size; (void)d_ws; (void)ws_size;

    const float* x       = (const float*)d_in[0];
    const float* Wenc    = (const float*)d_in[1];
    const float* benc    = (const float*)d_in[2];
    const float* Wdec    = (const float*)d_in[3];
    const float* prebias = (const float*)d_in[4];
    // d_in[5] is k==64, baked in as KK.

    float* out    = (float*)d_out;
    float* xhat   = out;                                   // B*H
    float* topv   = out + (size_t)BB * HH;                 // B*K
    float* topi   = topv + (size_t)BB * KK;                // B*K
    float* preact = topi + (size_t)BB * KK;                // B*D

    // Encoder: (B/128)*(D/128) = 8*512 blocks, M-index fastest for L2 reuse of W_enc.
    sae_encoder_wmma<<<dim3((BB / 128) * (DD / 128)), dim3(256), 0, stream>>>(
        x, Wenc, benc, prebias, preact);

    // Top-k: one 1024-thread block per row.
    sae_topk<<<dim3(BB), dim3(1024), 0, stream>>>(preact, topv, topi);

    // Decoder: one block per h, 256 KB dynamic LDS for the W_dec row (TDM-staged).
    sae_decoder<<<dim3(HH), dim3(256), DD * sizeof(float), stream>>>(
        Wdec, prebias, topv, topi, xhat);
}
